// get_model_15384572854600
// MI455X (gfx1250) — matrix operand order repaired and checked by reading, not compiled, not run on hardware
//
#include <hip/hip_runtime.h>
#include <hip/hip_bf16.h>

typedef __attribute__((ext_vector_type(16))) __bf16 v16bf;
typedef __attribute__((ext_vector_type(8)))  float  v8f;

#define DEV __device__ __forceinline__

constexpr int NN = 2048;   // points per cloud
constexpr int KNB = 40;    // kNN neighbors
constexpr int BB = 8;      // batch

DEV float lrelu_f(float v) { return v > 0.f ? v : 0.2f * v; }

DEV void atomicMaxF(float* addr, float val) {
  int old = __float_as_int(*addr);
  while (__int_as_float(old) < val) {
    int assumed = old;
    old = atomicCAS((int*)addr, assumed, __float_as_int(val));
    if (old == assumed) break;
  }
}

// ---------------------------------------------------------------------------
// Weight repack: A[M x Kdim] f32 row-major -> bf16 fragments in the exact
// v_wmma_f32_16x16x32_bf16 A-operand layout (ISA 7.12.2), zero-padded in both
// M (to Mtiles*16) and K (to ksteps*32).
// Apk[((mt*ksteps + ks)*32 + lane)*16 + e],  e = 2*v + h
//   lane: g=lane>>4, r=lane&15 ; row M = mt*16 + r
//   K = ks*32 + (v<4?0:16) + g*8 + 2*(v&3) + h
// ---------------------------------------------------------------------------
__global__ void repack_A(const float* __restrict__ A, int M, int Kdim,
                         int ksteps, int Mtiles, __bf16* __restrict__ Apk)
{
  int i = blockIdx.x * blockDim.x + threadIdx.x;
  int tot = Mtiles * ksteps * 512;
  if (i >= tot) return;
  int e = i & 15;
  int lane = (i >> 4) & 31;
  int t = i >> 9;
  int ks = t % ksteps;
  int mt = t / ksteps;
  int g = lane >> 4, r = lane & 15;
  int v = e >> 1, h = e & 1;
  int ka = ks * 32 + ((v < 4) ? 0 : 16) + g * 8 + 2 * (v & 3) + h;
  int mr = mt * 16 + r;
  float av = (ka < Kdim && mr < M) ? A[mr * Kdim + ka] : 0.f;
  Apk[i] = (__bf16)av;
}

// ---------------------------------------------------------------------------
// bf16 WMMA GEMM: D[Mpad x Ncols] = A(Apk fragments) * B[Kdim x Ncols]
// Block = 128 threads (4 waves); each block computes a 64x16 output tile:
// the 4 waves stack in M and SHARE one LDS-staged 32x16 B tile per K-step.
// B staging is branch-free (K clamped; zero-padded A annihilates overflow).
//   BMODE 0: plain   B[c,j] = Bact[b*bStride + c*Ncols + j]
//   BMODE 1: edge    j->(n=j/KNB,k=j%KNB), m=idx[b,n,k]
//                    c<Cin: x[b,c,m]-x[b,c,n] ; else x[b,c-Cin,n]
//   BMODE 2: bf16 4D with BN+lrelu folded: lrelu((y4d[b,c,j]-mean[c])*rs[c])
//   EMODE 0: atomic max into ymax[b,o,n] (+ channel sum/sumsq atomics)
//   EMODE 1: store bf16 4D (+ stats)
//   EMODE 2: store f32     (+ stats)
//   EMODE 3: transposed final store outT[b,j,o] (o < Mreal), no stats
// All flat indices fit in int32 (largest tensor ~42M elements).
// ---------------------------------------------------------------------------
template <int BMODE, int EMODE>
__global__ void wmma_gemm(
    const __bf16* __restrict__ Apk, int Mreal, int Kdim, int ksteps, int Ncols,
    const float* __restrict__ Bact, int bStride, int Cin,
    const int* __restrict__ idxN, const __bf16* __restrict__ y4dIn,
    const float* __restrict__ nmean, const float* __restrict__ nrs,
    float* __restrict__ ymax, __bf16* __restrict__ y4dOut,
    float* __restrict__ yout, float* __restrict__ outT, int outLd,
    float* __restrict__ ssum, float* __restrict__ ssq)
{
  __shared__ __align__(32) __bf16 btile[512];   // 32x16 B tile, fragment layout
  const int tid = threadIdx.x;
  const int wave = tid >> 5, lane = tid & 31;
  const int g = lane >> 4, r = lane & 15;
  const int mt = blockIdx.x * 4 + wave;
  const int j0 = blockIdx.y * 16;
  const int b = blockIdx.z;

  v8f acc;
#pragma unroll
  for (int i = 0; i < 8; i++) acc[i] = 0.f;

  const __bf16* aptr = Apk + mt * ksteps * 512 + lane * 16;

  for (int ks = 0; ks < ksteps; ks++) {
    // ---- cooperative, branch-free B staging: 512 elements / 128 threads ----
#pragma unroll
    for (int i = 0; i < 4; i++) {
      int e = tid + 128 * i;
      int kbl = e >> 4;            // 0..31 : K within step
      int rr = e & 15;             // column within tile
      int kb = ks * 32 + kbl;
      int kbc = kb < Kdim ? kb : Kdim - 1;   // clamp; A is zero there
      int j = j0 + rr;
      float bv;
      if (BMODE == 0) {
        bv = Bact[b * bStride + kbc * Ncols + j];
      } else if (BMODE == 1) {
        int n = j / KNB;                       // constant divisor -> mul/shift
        int k = j - n * KNB;
        int mi = idxN[(b * NN + n) * KNB + k];
        bv = (kbc < Cin)
                 ? Bact[b * bStride + kbc * NN + mi] -
                       Bact[b * bStride + kbc * NN + n]
                 : Bact[b * bStride + (kbc - Cin) * NN + n];
      } else {
        float t = (float)y4dIn[(b * Cin + kbc) * Ncols + j];
        t = (t - nmean[kbc]) * nrs[kbc];
        bv = t > 0.f ? t : 0.2f * t;
      }
      // fragment slot: dest lane = ((kbl>>3)&1)*16 + rr, element = (kbl&7) + (kbl>>4)*8
      btile[((((kbl >> 3) & 1) << 4) + rr) * 16 + ((kbl & 7) | ((kbl >> 4) << 3))] = (__bf16)bv;
    }
    __syncthreads();

    v16bf af = *(const v16bf*)(aptr + ks * 512);
    __builtin_prefetch((const void*)(aptr + (ks + 1) * 512), 0, 1);
    v16bf bfr = *(const v16bf*)&btile[lane * 16];
    acc = __builtin_amdgcn_wmma_f32_16x16x32_bf16(false, af, false, bfr,
                                                  (short)0, acc, false, false);
    __syncthreads();
  }

  // ---- epilogue: vgpr v holds row M = mt*16 + v + 8*g, col N = j0 + r ----
  const int m0 = mt * 16;
  const int j = j0 + r;
#pragma unroll
  for (int v = 0; v < 8; v++) {
    int o = m0 + v + 8 * g;
    float y = acc[v];
    if (EMODE == 0) {
      int n = j / KNB;
      atomicMaxF(&ymax[(b * Mreal + o) * NN + n], y);
    } else if (EMODE == 1) {
      y4dOut[(b * Mreal + o) * Ncols + j] = (__bf16)y;
    } else if (EMODE == 2) {
      yout[(b * Mreal + o) * Ncols + j] = y;
    } else {
      if (o < Mreal) outT[(b * Ncols + j) * outLd + o] = y;
    }
    if (EMODE != 3) {
      // channel o shared by the 16 lanes of a group for fixed v:
      // butterfly-reduce, one atomic pair per (tile, v, g)
      float s1 = y, s2 = y * y;
      for (int t = 1; t < 16; t <<= 1) {
        s1 += __shfl_xor(s1, t, 32);
        s2 += __shfl_xor(s2, t, 32);
      }
      if (r == 0) {
        atomicAdd(&ssum[o], s1);
        atomicAdd(&ssq[o], s2);
      }
    }
  }
}

// ---------------------------------------------------------------------------
// kNN: one wave per query point. Distances to all 2048 points in LDS, then
// 40 argmin passes with wave butterfly reduction (tie -> lowest index,
// matching jax.lax.top_k on -dist).
// ---------------------------------------------------------------------------
__global__ void knn_kernel(const float* __restrict__ x, int bStride, int C,
                           int* __restrict__ idxOut)
{
  __shared__ float dist[NN];
  __shared__ float q[64];
  const int bn = blockIdx.x;
  const int b = bn >> 11, n = bn & (NN - 1);
  const float* xb = x + b * bStride;
  const int l = threadIdx.x;

  for (int c = l; c < C; c += 32) q[c] = xb[c * NN + n];
  __syncthreads();

  for (int m = l; m < NN; m += 32) {
    float d = 0.f;
    for (int c = 0; c < C; c++) {
      float t = xb[c * NN + m] - q[c];
      d += t * t;
    }
    dist[m] = d;
  }
  __syncthreads();

  for (int kk = 0; kk < KNB; kk++) {
    float best = 3.4e38f;
    int bi = NN;
    for (int m = l; m < NN; m += 32) {
      float d = dist[m];
      if (d < best || (d == best && m < bi)) { best = d; bi = m; }
    }
    for (int s = 1; s < 32; s <<= 1) {
      float ob = __shfl_xor(best, s, 32);
      int oi = __shfl_xor(bi, s, 32);
      if (ob < best || (ob == best && oi < bi)) { best = ob; bi = oi; }
    }
    if (l == 0) {
      idxOut[bn * KNB + kk] = bi;
      dist[bi] = 3.4e38f;
    }
    __syncthreads();
  }
}

// ---------------------------------------------------------------------------
// Small helpers
// ---------------------------------------------------------------------------
__global__ void fill_f32(float* p, float v, int n) {
  int i = blockIdx.x * blockDim.x + threadIdx.x;
  if (i < n) p[i] = v;
}

__global__ void finalize_stats(const float* __restrict__ ssum,
                               const float* __restrict__ ssq, float cntInv,
                               int O, float* __restrict__ mean,
                               float* __restrict__ rs)
{
  int o = blockIdx.x * blockDim.x + threadIdx.x;
  if (o >= O) return;
  float m = ssum[o] * cntInv;
  float v = ssq[o] * cntInv - m * m;
  mean[o] = m;
  rs[o] = rsqrtf(v + 1e-5f);
}

// dst[b*dstBStride + o*NN + n] = lrelu((src[b,o,n]-mean[o])*rs[o])
__global__ void bn_act(const float* __restrict__ src, int O,
                       const float* __restrict__ mean,
                       const float* __restrict__ rs, float* __restrict__ dst,
                       int dstBStride)
{
  int i = blockIdx.x * blockDim.x + threadIdx.x;
  int tot = BB * O * NN;
  if (i >= tot) return;
  int n = i & (NN - 1);
  int t = i >> 11;
  int o = t % O;
  int b = t / O;
  float v = (src[i] - mean[o]) * rs[o];
  dst[b * dstBStride + o * NN + n] = lrelu_f(v);
}

// out[b*O+o] = lrelu((max_n src[b,o,n] - mean[o]) * rs[o])   (one wave per (b,o))
__global__ void bn_act_maxN(const float* __restrict__ src, int O,
                            const float* __restrict__ mean,
                            const float* __restrict__ rs,
                            float* __restrict__ out)
{
  int bo = blockIdx.x;
  const float* row = src + bo * NN;
  float mx = -3.4e38f;
  for (int n = threadIdx.x; n < NN; n += 32) mx = fmaxf(mx, row[n]);
  for (int s = 1; s < 32; s <<= 1) mx = fmaxf(mx, __shfl_xor(mx, s, 32));
  if (threadIdx.x == 0) {
    int o = bo % O;
    out[bo] = lrelu_f((mx - mean[o]) * rs[o]);
  }
}

__global__ void fc_kernel(const float* __restrict__ W,
                          const float* __restrict__ in, int Kd, int O,
                          float* __restrict__ out)
{
  int i = blockIdx.x * blockDim.x + threadIdx.x;
  if (i >= O * BB) return;
  int o = i % O, b = i / O;
  float s = 0.f;
  for (int c = 0; c < Kd; c++) s += W[o * Kd + c] * in[b * Kd + c];
  out[b * O + o] = s;
}

// BN across batch (B=8) per feature + lrelu
__global__ void bn_vec(const float* __restrict__ y, int O, float* __restrict__ out)
{
  int o = blockIdx.x * blockDim.x + threadIdx.x;
  if (o >= O) return;
  float m = 0.f;
  for (int b = 0; b < BB; b++) m += y[b * O + o];
  m *= 0.125f;
  float v = 0.f;
  for (int b = 0; b < BB; b++) {
    float d = y[b * O + o] - m;
    v += d * d;
  }
  v *= 0.125f;
  float r = rsqrtf(v + 1e-5f);
  for (int b = 0; b < BB; b++)
    out[b * O + o] = lrelu_f((y[b * O + o] - m) * r);
}

// t9[b, i] = t256[b,:] . tTW[i,:] + tTb[i]
__global__ void t9_kernel(const float* __restrict__ t256,
                          const float* __restrict__ tTW,
                          const float* __restrict__ tTb, float* __restrict__ t9)
{
  int i = threadIdx.x;
  if (i >= 72) return;
  int b = i / 9, jj = i % 9;
  float s = tTb[jj];
  for (int c = 0; c < 256; c++) s += t256[b * 256 + c] * tTW[jj * 256 + c];
  t9[b * 9 + jj] = s;
}

// xt[b,d,n] = sum_c x[b,c,n] * t9[b, c*3 + d]
__global__ void apply_transform(const float* __restrict__ x,
                                const float* __restrict__ t9,
                                float* __restrict__ xt)
{
  int i = blockIdx.x * blockDim.x + threadIdx.x;
  if (i >= BB * 3 * NN) return;
  int n = i & (NN - 1);
  int t = i >> 11;
  int d = t % 3;
  int b = t / 3;
  float s = 0.f;
  for (int c = 0; c < 3; c++)
    s += x[(b * 3 + c) * NN + n] * t9[b * 9 + c * 3 + d];
  xt[(b * 3 + d) * NN + n] = s;
}

// lf[b,o] = lrelu(bn_over_B(W7[o, l[b]]))
__global__ void label_feat(const float* __restrict__ W7,
                           const int* __restrict__ lab, float* __restrict__ lf)
{
  int o = threadIdx.x;
  if (o >= 64) return;
  float y[BB];
  float m = 0.f;
  for (int b = 0; b < BB; b++) {
    y[b] = W7[o * 16 + lab[b]];
    m += y[b];
  }
  m *= 0.125f;
  float v = 0.f;
  for (int b = 0; b < BB; b++) {
    float d = y[b] - m;
    v += d * d;
  }
  v *= 0.125f;
  float r = rsqrtf(v + 1e-5f);
  for (int b = 0; b < BB; b++) lf[b * 64 + o] = lrelu_f((y[b] - m) * r);
}

// h[b,c,n]: c<1024 -> g1024[b,c]; c<1088 -> lf[b,c-1024]; else xc[b,c-1088,n]
__global__ void build_h(const float* __restrict__ g1024,
                        const float* __restrict__ lf,
                        const float* __restrict__ xc, float* __restrict__ h)
{
  int i = blockIdx.x * blockDim.x + threadIdx.x;
  int tot = BB * 1344 * NN;
  if (i >= tot) return;
  int n = i & (NN - 1);
  int t = i >> 11;
  int c = t % 1344;
  int b = t / 1344;
  float v;
  if (c < 1024) v = g1024[b * 1024 + c];
  else if (c < 1088) v = lf[b * 64 + (c - 1024)];
  else v = xc[(b * 256 + (c - 1088)) * NN + n];
  h[i] = v;
}

// ---------------------------------------------------------------------------
extern "C" void kernel_launch(void* const* d_in, const int* in_sizes, int n_in,
                              void* d_out, int out_size, void* d_ws, size_t ws_size,
                              hipStream_t stream)
{
  (void)in_sizes; (void)n_in; (void)out_size; (void)ws_size;
  const int Nn = NN, KN = KNB;
  const int NK = Nn * KN;

  const float* x   = (const float*)d_in[0];
  const int*   lab = (const int*)d_in[1];
  const float* tW1 = (const float*)d_in[2];
  const float* tW2 = (const float*)d_in[3];
  const float* tW3 = (const float*)d_in[4];
  const float* tL1 = (const float*)d_in[5];
  const float* tL2 = (const float*)d_in[6];
  const float* tTW = (const float*)d_in[7];
  const float* tTb = (const float*)d_in[8];
  const float* W1  = (const float*)d_in[9];
  const float* W2  = (const float*)d_in[10];
  const float* W3  = (const float*)d_in[11];
  const float* W6  = (const float*)d_in[12];
  const float* W7  = (const float*)d_in[13];
  const float* W8  = (const float*)d_in[14];
  const float* W9  = (const float*)d_in[15];
  const float* W10 = (const float*)d_in[16];
  const float* W11 = (const float*)d_in[17];
  float* out = (float*)d_out;

  // ---- workspace carve-up ----
  size_t cur = 0;
  auto alloc = [&](size_t bytes) -> char* {
    cur = (cur + 255) & ~(size_t)255;
    char* p = (char*)d_ws + cur;
    cur += bytes;
    return p;
  };
  float*  xt     = (float*)alloc((size_t)BB * 3 * Nn * 4);
  int*    idx    = (int*)alloc((size_t)BB * Nn * KN * 4);
  __bf16* Apk    = (__bf16*)alloc((size_t)1024 * 1024);     // packed weights
  __bf16* y4d    = (__bf16*)alloc((size_t)BB * 64 * NK * 2);
  float*  ymax   = (float*)alloc((size_t)BB * 128 * Nn * 4);
  float*  ycv    = (float*)alloc((size_t)BB * 1024 * Nn * 4);
  float*  ssum   = (float*)alloc(1024 * 4);
  float*  ssq    = (float*)alloc(1024 * 4);
  float*  mean   = (float*)alloc(1024 * 4);
  float*  rs     = (float*)alloc(1024 * 4);
  float*  t128   = (float*)alloc((size_t)BB * 128 * Nn * 4);
  float*  tv1024 = (float*)alloc(BB * 1024 * 4);
  float*  tv512a = (float*)alloc(BB * 512 * 4);
  float*  tv512b = (float*)alloc(BB * 512 * 4);
  float*  tv256a = (float*)alloc(BB * 256 * 4);
  float*  tv256b = (float*)alloc(BB * 256 * 4);
  float*  t9     = (float*)alloc(BB * 9 * 4);
  float*  g1024  = (float*)alloc(BB * 1024 * 4);
  float*  lf     = (float*)alloc(BB * 64 * 4);
  float*  xc     = (float*)alloc((size_t)BB * 256 * Nn * 4);
  float*  hh     = (float*)alloc((size_t)BB * 1344 * Nn * 4);
  float*  a8     = (float*)alloc((size_t)BB * 256 * Nn * 4);
  float*  a9     = (float*)alloc((size_t)BB * 256 * Nn * 4);
  float*  a10    = (float*)alloc((size_t)BB * 128 * Nn * 4);

  auto fill = [&](float* p, float v, int n) {
    fill_f32<<<dim3((unsigned)((n + 255) / 256)), 256, 0, stream>>>(p, v, n);
  };
  auto clearStats = [&]() { fill(ssum, 0.f, 1024); fill(ssq, 0.f, 1024); };

  auto gemm = [&](const float* A, int M, int Kdim, int Ncols, int bmode,
                  const float* Bact, int bStride, int Cin,
                  const int* idxp, const __bf16* y4in,
                  int emode, float* ymaxp, __bf16* y4out, float* youtp,
                  float* outTp, int outLd) {
    int Mpad = (M + 63) & ~63;            // block covers 64 rows (4 waves)
    int Mtiles = Mpad / 16;
    int ksteps = (Kdim + 31) / 32;
    int tot = Mtiles * ksteps * 512;
    repack_A<<<dim3((unsigned)((tot + 255) / 256)), 256, 0, stream>>>(
        A, M, Kdim, ksteps, Mtiles, Apk);
    dim3 grid((unsigned)(Mpad / 64), (unsigned)(Ncols / 16), BB);
#define GEMM_ARGS Apk, M, Kdim, ksteps, Ncols, Bact, bStride, Cin, idxp, y4in, \
                  mean, rs, ymaxp, y4out, youtp, outTp, outLd, ssum, ssq
    if (bmode == 0 && emode == 2)      wmma_gemm<0, 2><<<grid, 128, 0, stream>>>(GEMM_ARGS);
    else if (bmode == 0 && emode == 3) wmma_gemm<0, 3><<<grid, 128, 0, stream>>>(GEMM_ARGS);
    else if (bmode == 1 && emode == 0) wmma_gemm<1, 0><<<grid, 128, 0, stream>>>(GEMM_ARGS);
    else if (bmode == 1 && emode == 1) wmma_gemm<1, 1><<<grid, 128, 0, stream>>>(GEMM_ARGS);
    else if (bmode == 2 && emode == 0) wmma_gemm<2, 0><<<grid, 128, 0, stream>>>(GEMM_ARGS);
#undef GEMM_ARGS
  };
  auto finalize = [&](int O, float cnt) {
    finalize_stats<<<dim3((unsigned)((O + 255) / 256)), 256, 0, stream>>>(
        ssum, ssq, 1.0f / cnt, O, mean, rs);
  };
  auto bnact = [&](const float* src, int O, float* dst, int dstBStride) {
    int tot = BB * O * Nn;
    bn_act<<<dim3((unsigned)((tot + 255) / 256)), 256, 0, stream>>>(
        src, O, mean, rs, dst, dstBStride);
  };

  const float cntNK = (float)BB * (float)NK;
  const float cntN = (float)BB * (float)Nn;

  // ================= Transform net =================
  knn_kernel<<<BB * Nn, 32, 0, stream>>>(x, 3 * Nn, 3, idx);
  clearStats();
  gemm(tW1, 64, 6, NK, 1, x, 3 * Nn, 3, idx, nullptr,
       1, nullptr, y4d, nullptr, nullptr, 0);
  finalize(64, cntNK);
  clearStats();
  fill(ymax, -3.0e38f, BB * 128 * Nn);
  gemm(tW2, 128, 64, NK, 2, nullptr, 0, 64, nullptr, y4d,
       0, ymax, nullptr, nullptr, nullptr, 0);
  finalize(128, cntNK);
  bnact(ymax, 128, t128, 128 * Nn);
  clearStats();
  gemm(tW3, 1024, 128, Nn, 0, t128, 128 * Nn, 0, nullptr, nullptr,
       2, nullptr, nullptr, ycv, nullptr, 0);
  finalize(1024, cntN);
  bn_act_maxN<<<BB * 1024, 32, 0, stream>>>(ycv, 1024, mean, rs, tv1024);
  fc_kernel<<<16, 256, 0, stream>>>(tL1, tv1024, 1024, 512, tv512a);
  bn_vec<<<2, 256, 0, stream>>>(tv512a, 512, tv512b);
  fc_kernel<<<8, 256, 0, stream>>>(tL2, tv512b, 512, 256, tv256a);
  bn_vec<<<1, 256, 0, stream>>>(tv256a, 256, tv256b);
  t9_kernel<<<1, 128, 0, stream>>>(tv256b, tTW, tTb, t9);
  apply_transform<<<(BB * 3 * Nn + 255) / 256, 256, 0, stream>>>(x, t9, xt);

  // ================= Main branch =================
  // x1
  knn_kernel<<<BB * Nn, 32, 0, stream>>>(xt, 3 * Nn, 3, idx);
  clearStats();
  fill(ymax, -3.0e38f, BB * 64 * Nn);
  gemm(W1, 64, 6, NK, 1, xt, 3 * Nn, 3, idx, nullptr,
       0, ymax, nullptr, nullptr, nullptr, 0);
  finalize(64, cntNK);
  bnact(ymax, 64, xc + 0, 256 * Nn);
  // x2
  knn_kernel<<<BB * Nn, 32, 0, stream>>>(xc, 256 * Nn, 64, idx);
  clearStats();
  fill(ymax, -3.0e38f, BB * 64 * Nn);
  gemm(W2, 64, 128, NK, 1, xc, 256 * Nn, 64, idx, nullptr,
       0, ymax, nullptr, nullptr, nullptr, 0);
  finalize(64, cntNK);
  bnact(ymax, 64, xc + 64 * Nn, 256 * Nn);
  // x3
  knn_kernel<<<BB * Nn, 32, 0, stream>>>(xc + 64 * Nn, 256 * Nn, 64, idx);
  clearStats();
  fill(ymax, -3.0e38f, BB * 128 * Nn);
  gemm(W3, 128, 128, NK, 1, xc + 64 * Nn, 256 * Nn, 64, idx, nullptr,
       0, ymax, nullptr, nullptr, nullptr, 0);
  finalize(128, cntNK);
  bnact(ymax, 128, xc + 128 * Nn, 256 * Nn);
  // global feature
  clearStats();
  gemm(W6, 1024, 256, Nn, 0, xc, 256 * Nn, 0, nullptr, nullptr,
       2, nullptr, nullptr, ycv, nullptr, 0);
  finalize(1024, cntN);
  bn_act_maxN<<<BB * 1024, 32, 0, stream>>>(ycv, 1024, mean, rs, g1024);
  // label feature
  label_feat<<<1, 64, 0, stream>>>(W7, lab, lf);
  // concat h = [g, lf, x1, x2, x3]
  {
    int tot = BB * 1344 * Nn;
    build_h<<<dim3((unsigned)((tot + 255) / 256)), 256, 0, stream>>>(g1024, lf, xc, hh);
  }
  // W8 / W9 / W10
  clearStats();
  gemm(W8, 256, 1344, Nn, 0, hh, 1344 * Nn, 0, nullptr, nullptr,
       2, nullptr, nullptr, ycv, nullptr, 0);
  finalize(256, cntN);
  bnact(ycv, 256, a8, 256 * Nn);
  clearStats();
  gemm(W9, 256, 256, Nn, 0, a8, 256 * Nn, 0, nullptr, nullptr,
       2, nullptr, nullptr, ycv, nullptr, 0);
  finalize(256, cntN);
  bnact(ycv, 256, a9, 256 * Nn);
  clearStats();
  gemm(W10, 128, 256, Nn, 0, a9, 256 * Nn, 0, nullptr, nullptr,
       2, nullptr, nullptr, ycv, nullptr, 0);
  finalize(128, cntN);
  bnact(ycv, 128, a10, 128 * Nn);
  // W11: final logits, transposed store [B, N, 50]
  gemm(W11, 50, 128, Nn, 0, a10, 128 * Nn, 0, nullptr, nullptr,
       3, nullptr, nullptr, nullptr, out, 50);
}
